// IDCNN_29111288332735
// MI455X (gfx1250) — matrix operand order repaired and checked by reading, not compiled, not run on hardware
//
#include <hip/hip_runtime.h>

typedef unsigned int u32;
typedef unsigned short u16;

typedef __attribute__((ext_vector_type(16))) __bf16 v16bf;
typedef __attribute__((ext_vector_type(8)))  float  v8f;
typedef __attribute__((ext_vector_type(4)))  int    v4i;

// Problem constants (fixed by the reference)
constexpr int Bb = 128;   // batch
constexpr int L  = 400;   // sequence length
constexpr int D  = 1024;  // input size
constexpr int C  = 256;   // filters
constexpr float EPS = 1e-6f;

// Conv tiling
constexpr int LT     = 80;            // L-chunk per block (400 = 5*80)
constexpr int NCHUNK = L / LT;        // 5
constexpr int NT     = LT / 16;       // 5 n-tiles per block
constexpr int HALO   = 2;             // covers dilation 2
constexpr int LROWS  = LT + 2 * HALO; // 84
constexpr int CPAD   = C + 8;         // LDS row pad (16B) to spread banks
constexpr int DPAD   = D + 8;         // LDS row pad for linear staging

#if defined(__HIP_DEVICE_COMPILE__) && __has_builtin(__builtin_amdgcn_global_load_async_to_lds_b128)
#define USE_ASYNC_LDS 1
#endif

// Native hardware converts (v_cvt_*bf16_f32 / shift) instead of manual RNE bit math
__device__ __forceinline__ u16 f2bf(float f) {
    union { __bf16 h; u16 u; } c;
    c.h = (__bf16)f;
    return c.u;
}
__device__ __forceinline__ float bf2f(u16 h) {
    union { u16 u; __bf16 h; } c;
    c.u = h;
    return (float)c.h;
}

union Frag {            // one WMMA 16-bit operand: 16 bf16 = 32 bytes = 8 VGPRs
    v16bf v;
    uint4 q[2];
    u16   s[16];
};

// ---------------------------------------------------------------------------
// Weight prep: fp32 -> bf16 in fragment-friendly layouts
// ---------------------------------------------------------------------------
__global__ __launch_bounds__(256) void prep_wl(const float* __restrict__ Wl,
                                               u16* __restrict__ wlT) {
    // Wl [D][C] -> wlT [C][D]
    int i = blockIdx.x * 256 + threadIdx.x;
    if (i >= D * C) return;
    int c = i / D, d = i - c * D;
    wlT[(size_t)c * D + d] = f2bf(Wl[(size_t)d * C + c]);
}

__global__ __launch_bounds__(256) void prep_w(const float* __restrict__ w,
                                              u16* __restrict__ wt) {
    // w [C_out][C_in][3] (OIH) -> wt [3][C_out][C_in]
    int i = blockIdx.x * 256 + threadIdx.x;
    if (i >= 3 * C * C) return;
    int tap = i / (C * C);
    int rem = i - tap * C * C;
    int co = rem / C, ci = rem - co * C;
    wt[i] = f2bf(w[((size_t)co * C + ci) * 3 + tap]);
}

// ---------------------------------------------------------------------------
// Shared epilogue: bias (+optional relu), store bf16 or fp32
// ---------------------------------------------------------------------------
__device__ __forceinline__ void store_tile(v8f acc, const float* __restrict__ bias,
                                           int cb, void* __restrict__ outp,
                                           size_t rowoff, int do_relu, int outFp32) {
    float4 b0 = *(const float4*)(bias + cb);
    float4 b1 = *(const float4*)(bias + cb + 4);
    float r[8] = {acc[0] + b0.x, acc[1] + b0.y, acc[2] + b0.z, acc[3] + b0.w,
                  acc[4] + b1.x, acc[5] + b1.y, acc[6] + b1.z, acc[7] + b1.w};
    if (do_relu) {
#pragma unroll
        for (int i = 0; i < 8; ++i) r[i] = fmaxf(r[i], 0.f);
    }
    if (outFp32) {
        float* o = (float*)outp + rowoff + cb;
        *(float4*)(o)     = make_float4(r[0], r[1], r[2], r[3]);
        *(float4*)(o + 4) = make_float4(r[4], r[5], r[6], r[7]);
    } else {
        union { uint4 q; u16 s[8]; } pk;
#pragma unroll
        for (int i = 0; i < 8; ++i) pk.s[i] = f2bf(r[i]);
        *(uint4*)((u16*)outp + rowoff + cb) = pk.q;
    }
}

// ---------------------------------------------------------------------------
// Linear: out[b][l][c] = bf16( emb[b][l][:] @ Wl[:, c] + bl[c] )
// emb tile staged once per block into LDS as bf16 (converted once).
// ---------------------------------------------------------------------------
__global__ __launch_bounds__(256) void linear_wmma(const float* __restrict__ emb,
                                                   const u16* __restrict__ wlT,
                                                   const float* __restrict__ bl,
                                                   u16* __restrict__ out) {
    __shared__ u16 es[16][DPAD];

    const int blk = blockIdx.x;
    const int b = blk / (L / 16), lt = blk % (L / 16);
    const int l0 = lt * 16;
    const int tid = threadIdx.x;

    // Stage emb[b][l0..l0+16)[0..D) -> bf16 LDS (8 floats per chunk)
    {
        const float* src = emb + ((size_t)b * L + l0) * D;
        for (int f = tid; f < 16 * (D / 8); f += 256) {
            int row = f >> 7;                 // D/8 = 128 chunks per row
            int c8 = (f & 127) * 8;
            const float4* p = (const float4*)(src + (size_t)row * D + c8);
            float4 x0 = p[0], x1 = p[1];
            union { uint4 q; u16 s[8]; } pk;
            pk.s[0] = f2bf(x0.x); pk.s[1] = f2bf(x0.y);
            pk.s[2] = f2bf(x0.z); pk.s[3] = f2bf(x0.w);
            pk.s[4] = f2bf(x1.x); pk.s[5] = f2bf(x1.y);
            pk.s[6] = f2bf(x1.z); pk.s[7] = f2bf(x1.w);
            *(uint4*)&es[row][c8] = pk.q;
        }
    }
    __syncthreads();

    const int wave = tid >> 5, lane = tid & 31;
    const int n = lane & 15, half = lane >> 4, mrow = lane & 15;
    const int c00 = wave * 16, c01 = (wave + 8) * 16;

    v8f acc0 = {}, acc1 = {};
    const u16* brow = &es[n][0];
    const u16* w0r = wlT + (size_t)(c00 + mrow) * D;
    const u16* w1r = wlT + (size_t)(c01 + mrow) * D;

    for (int d0 = 0; d0 < D; d0 += 32) {
        Frag bfr, a0, a1;
        bfr.q[0] = *(const uint4*)(brow + d0 + 8 * half);
        bfr.q[1] = *(const uint4*)(brow + d0 + 16 + 8 * half);
        a0.q[0] = *(const uint4*)(w0r + d0 + 8 * half);
        a0.q[1] = *(const uint4*)(w0r + d0 + 16 + 8 * half);
        a1.q[0] = *(const uint4*)(w1r + d0 + 8 * half);
        a1.q[1] = *(const uint4*)(w1r + d0 + 16 + 8 * half);

        acc0 = __builtin_amdgcn_wmma_f32_16x16x32_bf16(false, a0.v, false, bfr.v,
                                                       (short)0, acc0, false, false);
        acc1 = __builtin_amdgcn_wmma_f32_16x16x32_bf16(false, a1.v, false, bfr.v,
                                                       (short)0, acc1, false, false);
    }
    size_t rowoff = ((size_t)b * L + (l0 + n)) * C;
    store_tile(acc0, bl, c00 + 8 * half, out, rowoff, 0, 0);
    store_tile(acc1, bl, c01 + 8 * half, out, rowoff, 0, 0);
}

// ---------------------------------------------------------------------------
// Conv1d (K=3, pad=dil) as 3 tap-GEMMs on WMMA.
// LDS halo tile loaded via async global->LDS (ASYNCcnt path) when available.
// Each wave: 2 m-tiles x 5 n-tiles, 10 accumulators -> 10 independent WMMAs
// per (tap,ck) step (no RAW chain), A loaded once per step.
//   xin: bf16 [B][L][C]   wt: bf16 [3][C_out][C_in]
// ---------------------------------------------------------------------------
__global__ __launch_bounds__(256) void conv_wmma(const u16* __restrict__ xin,
                                                 const u16* __restrict__ wt,
                                                 const float* __restrict__ bias,
                                                 void* __restrict__ outp,
                                                 int dil, int do_relu, int outFp32) {
    __shared__ u16 xs[LROWS][CPAD];

    const int blk = blockIdx.x;
    const int b = blk / NCHUNK, chunk = blk % NCHUNK;
    const int l0 = chunk * LT;
    const int tid = threadIdx.x;

#ifdef USE_ASYNC_LDS
    {
        // Zero only the out-of-sequence halo rows (never touched by async copy)
        if (l0 == 0) {
            for (int f = tid; f < HALO * (C / 2); f += 256) {
                int row = f / (C / 2), col = f % (C / 2);
                *((u32*)&xs[row][0] + col) = 0u;
            }
        }
        if (l0 + LT == L) {
            for (int f = tid; f < HALO * (C / 2); f += 256) {
                int row = LROWS - HALO + f / (C / 2), col = f % (C / 2);
                *((u32*)&xs[row][0] + col) = 0u;
            }
        }
        const int r0 = (l0 == 0) ? HALO : 0;
        const int r1 = (l0 + LT == L) ? (LROWS - HALO) : LROWS;
        const char* gsrc = (const char*)(xin + (size_t)b * L * C);
        const int total = (r1 - r0) * (C / 8);     // 16B chunks per row = 32
        for (int f = tid; f < total; f += 256) {
            int row = r0 + (f >> 5), cq = f & 31;
            uintptr_t g = (uintptr_t)(gsrc + (size_t)(l0 + row - HALO) * (C * 2) + cq * 16);
            __builtin_amdgcn_global_load_async_to_lds_b128(
                (__attribute__((address_space(1))) v4i*)g,
                (__attribute__((address_space(3))) v4i*)&xs[row][cq * 8], 0, 0);
        }
        asm volatile("s_wait_asynccnt 0" ::: "memory");
    }
#else
    {
        const u32* src = (const u32*)(xin + (size_t)b * L * C);
        const int W2 = C / 2;
        for (int f = tid; f < LROWS * W2; f += 256) {
            int row = f / W2, col = f - row * W2;
            int l = l0 + row - HALO;
            u32 v = 0u;
            if (l >= 0 && l < L) v = src[(size_t)l * W2 + col];
            *((u32*)&xs[row][0] + col) = v;
        }
    }
#endif
    __syncthreads();

    const int wave = tid >> 5, lane = tid & 31;
    const int n = lane & 15, half = lane >> 4, mrow = lane & 15;
    const int c0a = wave * 16, c0b = (wave + 8) * 16;

    v8f acc[2 * NT];
#pragma unroll
    for (int i = 0; i < 2 * NT; ++i) acc[i] = {};

    const u16* wrowa = wt + (size_t)(c0a + mrow) * C;
    const u16* wrowb = wt + (size_t)(c0b + mrow) * C;

    for (int tap = 0; tap < 3; ++tap) {
        const int roff = n + (tap - 1) * dil + HALO;  // LDS row offset within n-tile
        const u16* wa = wrowa + (size_t)tap * C * C;
        const u16* wb = wrowb + (size_t)tap * C * C;
        for (int ck = 0; ck < C; ck += 32) {
            Frag a0, a1;
            a0.q[0] = *(const uint4*)(wa + ck + 8 * half);
            a0.q[1] = *(const uint4*)(wa + ck + 16 + 8 * half);
            a1.q[0] = *(const uint4*)(wb + ck + 8 * half);
            a1.q[1] = *(const uint4*)(wb + ck + 16 + 8 * half);
#pragma unroll
            for (int nt = 0; nt < NT; ++nt) {
                Frag bfr;
                const u16* xrow = &xs[nt * 16 + roff][0];
                bfr.q[0] = *(const uint4*)(xrow + ck + 8 * half);
                bfr.q[1] = *(const uint4*)(xrow + ck + 16 + 8 * half);
                acc[2 * nt] = __builtin_amdgcn_wmma_f32_16x16x32_bf16(
                    false, a0.v, false, bfr.v, (short)0, acc[2 * nt], false, false);
                acc[2 * nt + 1] = __builtin_amdgcn_wmma_f32_16x16x32_bf16(
                    false, a1.v, false, bfr.v, (short)0, acc[2 * nt + 1], false, false);
            }
        }
    }

#pragma unroll
    for (int nt = 0; nt < NT; ++nt) {
        int lg = l0 + nt * 16 + n;
        size_t rowoff = ((size_t)b * L + lg) * C;
        store_tile(acc[2 * nt],     bias, c0a + 8 * half, outp, rowoff, do_relu, outFp32);
        store_tile(acc[2 * nt + 1], bias, c0b + 8 * half, outp, rowoff, do_relu, outFp32);
    }
}

// ---------------------------------------------------------------------------
// LayerNorm over L (ddof=1, /(std+eps)), in-place on bf16 [B][L][C].
// One block per batch, one thread per channel column; coalesced row sweeps.
// ---------------------------------------------------------------------------
__global__ __launch_bounds__(256) void ln_kernel(u16* __restrict__ x,
                                                 const float* __restrict__ g,
                                                 const float* __restrict__ be) {
    int b = blockIdx.x;
    int c = threadIdx.x;
    u16* col = x + (size_t)b * L * C + c;
    float s = 0.f, ss = 0.f;
    for (int l = 0; l < L; ++l) {
        float v = bf2f(col[(size_t)l * C]);
        s += v; ss += v * v;
    }
    float mean = s / (float)L;
    float var = (ss - (float)L * mean * mean) / (float)(L - 1);
    var = fmaxf(var, 0.f);
    float inv = 1.f / (sqrtf(var) + EPS);
    for (int l = 0; l < L; ++l) {
        float v = bf2f(col[(size_t)l * C]);
        float o = g[l] * (v - mean) * inv + be[l];
        col[(size_t)l * C] = f2bf(o);
    }
}

// ---------------------------------------------------------------------------
// Host orchestration
// ---------------------------------------------------------------------------
static inline size_t align256(size_t x) { return (x + 255) & ~(size_t)255; }

extern "C" void kernel_launch(void* const* d_in, const int* in_sizes, int n_in,
                              void* d_out, int out_size, void* d_ws, size_t ws_size,
                              hipStream_t stream) {
    (void)in_sizes; (void)n_in; (void)out_size; (void)ws_size;
    const float* emb = (const float*)d_in[0];
    // d_in[1] = length (unused)
    const float* Wl = (const float*)d_in[2];
    const float* bl = (const float*)d_in[3];
    const float* w0 = (const float*)d_in[4];
    const float* b0 = (const float*)d_in[5];
    const float* w1 = (const float*)d_in[6];
    const float* b1 = (const float*)d_in[7];
    const float* w2 = (const float*)d_in[8];
    const float* b2 = (const float*)d_in[9];
    const float* g1  = (const float*)d_in[10];
    const float* be1 = (const float*)d_in[11];
    const float* g2  = (const float*)d_in[12];
    const float* be2 = (const float*)d_in[13];

    char* p = (char*)d_ws;
    u16* wlT = (u16*)p;  p += align256((size_t)C * D * 2);
    u16* wt  = (u16*)p;  p += align256((size_t)9 * C * C * 2);  // 3 convs x [3][C][C]
    u16* bufA = (u16*)p; p += align256((size_t)Bb * L * C * 2);
    u16* bufT = (u16*)p;
    u16* wt0 = wt;
    u16* wt1 = wt + (size_t)3 * C * C;
    u16* wt2 = wt + (size_t)6 * C * C;

    // Weight prep (cheap, deterministic each call)
    prep_wl<<<(D * C + 255) / 256, 256, 0, stream>>>(Wl, wlT);
    prep_w<<<(3 * C * C + 255) / 256, 256, 0, stream>>>(w0, wt0);
    prep_w<<<(3 * C * C + 255) / 256, 256, 0, stream>>>(w1, wt1);
    prep_w<<<(3 * C * C + 255) / 256, 256, 0, stream>>>(w2, wt2);

    // h = emb @ Wl + bl  (stored [B][L][C] bf16 == transposed [B][C][L] view)
    linear_wmma<<<Bb * (L / 16), 256, 0, stream>>>(emb, wlT, bl, bufA);

    const int GC = Bb * NCHUNK;
    // ---- net 1 + outer relu/LN ----
    conv_wmma<<<GC, 256, 0, stream>>>(bufA, wt0, b0, bufT, 1, 1, 0);  // conv0 + relu
    ln_kernel<<<Bb, 256, 0, stream>>>(bufT, g1, be1);                 // inner LN
    conv_wmma<<<GC, 256, 0, stream>>>(bufT, wt1, b1, bufA, 1, 0, 0);  // conv1
    conv_wmma<<<GC, 256, 0, stream>>>(bufA, wt2, b2, bufT, 2, 1, 0);  // conv2 + outer relu
    ln_kernel<<<Bb, 256, 0, stream>>>(bufT, g2, be2);                 // outer LN
    // ---- net 2 ----
    conv_wmma<<<GC, 256, 0, stream>>>(bufT, wt0, b0, bufA, 1, 1, 0);
    ln_kernel<<<Bb, 256, 0, stream>>>(bufA, g1, be1);
    conv_wmma<<<GC, 256, 0, stream>>>(bufA, wt1, b1, bufT, 1, 0, 0);
    conv_wmma<<<GC, 256, 0, stream>>>(bufT, wt2, b2, bufA, 2, 0, 0);
    // ---- net 3 ----
    conv_wmma<<<GC, 256, 0, stream>>>(bufA, wt0, b0, bufT, 1, 1, 0);
    ln_kernel<<<Bb, 256, 0, stream>>>(bufT, g1, be1);
    conv_wmma<<<GC, 256, 0, stream>>>(bufT, wt1, b1, bufA, 1, 0, 0);
    conv_wmma<<<GC, 256, 0, stream>>>(bufA, wt2, b2, bufT, 2, 0, 0);
    // ---- net 4 (final conv2 writes fp32 to d_out, [B][L][C] = reference output) ----
    conv_wmma<<<GC, 256, 0, stream>>>(bufT, wt0, b0, bufA, 1, 1, 0);
    ln_kernel<<<Bb, 256, 0, stream>>>(bufA, g1, be1);
    conv_wmma<<<GC, 256, 0, stream>>>(bufA, wt1, b1, bufT, 1, 0, 0);
    conv_wmma<<<GC, 256, 0, stream>>>(bufT, wt2, b2, d_out, 2, 0, 1);
}
